// AtenMatmulQint8VM_18940805775749
// MI455X (gfx1250) — 8-bit matrix operand order repaired and checked by reading, compiled, not run on hardware
//
#include <hip/hip_runtime.h>


typedef int v8i __attribute__((ext_vector_type(8)));

#define Q_X_SCALE 0.0215
#define Q_X_ZP    (-25)
#define Q_Y_SCALE 0.0176
#define Q_Y_ZP    18

#define K_FIXED   8192
#define N_FIXED   16384
#define KSPLIT    4
#define KSEG      (K_FIXED / KSPLIT)
#define CHUNKS    (KSEG / 64)

#define WAVE            32
#define WAVES_PER_BLOCK 8
#define BLOCK           (WAVE * WAVES_PER_BLOCK)
#define COLS_PER_WAVE   16
#define COLS_PER_BLOCK  (WAVES_PER_BLOCK * COLS_PER_WAVE)

__device__ __forceinline__ uint32_t pack4(int a, int b, int c, int d) {
  return (uint32_t)(a & 0xff) | ((uint32_t)(b & 0xff) << 8) |
         ((uint32_t)(c & 0xff) << 16) | ((uint32_t)(d & 0xff) << 24);
}

template <int ROW_OFS>
__device__ __forceinline__ v8i load_b(const int* __restrict__ p) {
  v8i B;
#pragma unroll
  for (int v = 0; v < 8; ++v) {
    const int r = ROW_OFS + ((v >= 4) ? 32 : 0) + 4 * (v & 3);
    B[v] = (int)pack4(p[(size_t)(r + 0) * N_FIXED],
                      p[(size_t)(r + 1) * N_FIXED],
                      p[(size_t)(r + 2) * N_FIXED],
                      p[(size_t)(r + 3) * N_FIXED]);
  }
  return B;
}

__device__ __forceinline__ v8i load_a(const uint32_t* a_ptr) {
  const uint4 lo = *reinterpret_cast<const uint4*>(a_ptr);
  const uint4 hi = *reinterpret_cast<const uint4*>(a_ptr + 4);
  v8i A;
  A[0] = (int)lo.x; A[1] = (int)lo.y; A[2] = (int)lo.z; A[3] = (int)lo.w;
  A[4] = (int)hi.x; A[5] = (int)hi.y; A[6] = (int)hi.z; A[7] = (int)hi.w;
  return A;
}

__global__ void zero_ws_kernel(int* __restrict__ ws, int count) {
  int i = blockIdx.x * blockDim.x + threadIdx.x;
  if (i < count) ws[i] = 0;
}

__global__ __launch_bounds__(BLOCK) void qvm_wmma_kernel(
    const int* __restrict__ x, const int* __restrict__ y,
    int* __restrict__ ws_dot, int* __restrict__ ws_col) {
  const int seg_base = blockIdx.y * KSEG;
  const int tid  = threadIdx.x;

  __shared__ uint32_t lds_a[CHUNKS * 256];
  for (int i = tid; i < CHUNKS * 256; i += BLOCK) {
    const int c  = i >> 8;
    const int ln = (i >> 3) & 31;
    const int v  = i & 7;
    const int mm = ln & 15;
    const int hh = ln >> 4;
    uint32_t val = 0u;
    if (mm == 1) val = 0x01010101u;
    if (mm == 0) {
      const int dw = c * 16 + 8 * (v >> 2) + 4 * hh + (v & 3);
      const int4 xv = *reinterpret_cast<const int4*>(x + seg_base + 4 * dw);
      val = pack4(xv.x, xv.y, xv.z, xv.w);
    }
    lds_a[i] = val;
  }
  __syncthreads();

  const int wave = tid / WAVE;
  const int lane = tid % WAVE;
  const int m    = lane & 15;
  const int hi   = lane >> 4;
  const int ncol = blockIdx.x * COLS_PER_BLOCK + wave * COLS_PER_WAVE + m;

  const uint32_t* a_lane = &lds_a[(size_t)lane * 8];
  const int* p = y + (size_t)(seg_base + 16 * hi) * N_FIXED + (size_t)ncol;

  v8i acc0 = {};
  v8i acc1 = {};

  for (int c = 0; c < CHUNKS; c += 2) {
    if (c + 4 < CHUNKS) {
      __builtin_prefetch(p + (size_t)256 * N_FIXED, 0, 0);
    }

    const v8i A0 = load_a(a_lane + (size_t)(c + 0) * 256);
    const v8i B0 = load_b<0>(p);
    acc0 = __builtin_amdgcn_wmma_i32_16x16x64_iu8(
         false, A0,  false, B0, acc0,
         false,  false);

    const v8i A1 = load_a(a_lane + (size_t)(c + 1) * 256);
    const v8i B1 = load_b<64>(p);
    acc1 = __builtin_amdgcn_wmma_i32_16x16x64_iu8(
         false, A1,  false, B1, acc1,
         false,  false);

    p += (size_t)128 * N_FIXED;
  }

  if (lane < 16) {
    atomicAdd(&ws_dot[ncol], acc0[0] + acc1[0]);
    atomicAdd(&ws_col[ncol], acc0[1] + acc1[1]);
  }
}

__global__ __launch_bounds__(256) void finalize_kernel(
    const int* __restrict__ x, const int* __restrict__ ws_dot,
    const int* __restrict__ ws_col, float* __restrict__ out, int K, int N) {
  __shared__ int red[256];
  int s = 0;
  for (int k = threadIdx.x; k < K; k += 256) s += x[k];
  red[threadIdx.x] = s;
  __syncthreads();
  for (int off = 128; off > 0; off >>= 1) {
    if (threadIdx.x < off) red[threadIdx.x] += red[threadIdx.x + off];
    __syncthreads();
  }
  const long long sumx = red[0];
  const double S = Q_X_SCALE * Q_Y_SCALE;
  for (int n = threadIdx.x; n < N; n += 256) {
    const long long term = (long long)ws_dot[n]
                         - (long long)Q_Y_ZP * sumx
                         - (long long)Q_X_ZP * (long long)ws_col[n]
                         + (long long)K * (long long)Q_X_ZP * (long long)Q_Y_ZP;
    out[n] = (float)(S * (double)term);
  }
}

extern "C" void kernel_launch(void* const* d_in, const int* in_sizes, int n_in,
                              void* d_out, int out_size, void* d_ws, size_t ws_size,
                              hipStream_t stream) {
  const int* x = (const int*)d_in[0];
  const int* y = (const int*)d_in[1];
  const int  K = in_sizes[0];
  const int  N = in_sizes[1] / K;
  float* out   = (float*)d_out;

  int* ws_dot = (int*)d_ws;
  int* ws_col = ws_dot + N;

  zero_ws_kernel<<<(2 * N + 255) / 256, 256, 0, stream>>>(ws_dot, 2 * N);

  dim3 grid(N_FIXED / COLS_PER_BLOCK, KSPLIT);
  qvm_wmma_kernel<<<grid, BLOCK, 0, stream>>>(x, y, ws_dot, ws_col);

  finalize_kernel<<<1, 256, 0, stream>>>(x, ws_dot, ws_col, out, K, N);
}
